// GAT_9216999817292
// MI455X (gfx1250) — matrix operand order repaired and checked by reading, not compiled, not run on hardware
//
#include <hip/hip_runtime.h>
#include <hip/hip_bf16.h>
#include <math.h>

// ---------------- problem constants (from reference) ----------------
#define N_SRC 65536
#define N_DST 16384
#define NE    262144
#define DIN   256
#define HIDD  128
#define NH    4
#define KCAP  50
#define NEG_SLOPE 0.2f

// ---------------- CDNA5 WMMA vector types ----------------
typedef __attribute__((ext_vector_type(16))) __bf16 v16bf;
typedef __attribute__((ext_vector_type(8)))  __bf16 v8bf;
typedef __attribute__((ext_vector_type(8)))  float  v8f;

// ---------------- gfx1250 async global->LDS copy (ASYNCcnt) ----------------
#if defined(__has_builtin)
#if __has_builtin(__builtin_amdgcn_global_load_async_to_lds_b128)
#define HAVE_ASYNC_LDS 1
#endif
#if __has_builtin(__builtin_amdgcn_s_wait_asynccnt)
#define HAVE_WAIT_ASYNC 1
#endif
#endif

// builtin signature (from hipcc diagnostic): (v4i AS1*, v4i AS3*, imm int, imm int)
typedef int v4i_ __attribute__((vector_size(4 * sizeof(int))));
typedef __attribute__((address_space(1))) v4i_ as1_v4i;
typedef __attribute__((address_space(3))) v4i_ as3_v4i;

__device__ __forceinline__ void async_copy_b128(const unsigned short* g, unsigned short* l) {
#ifdef HAVE_ASYNC_LDS
  __builtin_amdgcn_global_load_async_to_lds_b128((as1_v4i*)g, (as3_v4i*)l, 0, 0);
#else
  *(uint4*)l = *(const uint4*)g;
#endif
}

__device__ __forceinline__ void wait_async_lds() {
#ifdef HAVE_ASYNC_LDS
#ifdef HAVE_WAIT_ASYNC
  __builtin_amdgcn_s_wait_asynccnt(0);
#else
  asm volatile("s_wait_asynccnt 0x0" ::: "memory");
#endif
#endif
}

__device__ __forceinline__ unsigned short f32_bf16(float f) {
  union { float f; unsigned u; } v; v.f = f;
  unsigned r = v.u + 0x7FFFu + ((v.u >> 16) & 1u);   // round-to-nearest-even
  return (unsigned short)(r >> 16);
}

// =====================================================================
// f32 -> bf16 row conversion (optionally gathering rows). 8 elems/thread.
// =====================================================================
__global__ void f32_to_bf16_rows(const float* __restrict__ src,
                                 unsigned short* __restrict__ dst,
                                 int rows, int cols,
                                 const int* __restrict__ gather)
{
  const int cpr = cols >> 3;                 // 16B chunks per row
  int i = blockIdx.x * blockDim.x + threadIdx.x;
  if (i >= rows * cpr) return;
  const int r  = i / cpr;
  const int ch = (i - r * cpr) * 8;
  const int sr = gather ? gather[r] : r;
  const float* sp = src + (size_t)sr * cols + ch;
  float4 x0 = *(const float4*)sp;
  float4 x1 = *(const float4*)(sp + 4);
  unsigned short o[8] = { f32_bf16(x0.x), f32_bf16(x0.y), f32_bf16(x0.z), f32_bf16(x0.w),
                          f32_bf16(x1.x), f32_bf16(x1.y), f32_bf16(x1.z), f32_bf16(x1.w) };
  *(uint4*)(dst + (size_t)r * cols + ch) = *(const uint4*)o;
}

// =====================================================================
// GEMM: C[M,N] = A[M,K](bf16) . W[N,K](bf16)^T, f32 accumulate.
// Block tile 128x64, BK=64, 256 threads (8 waves), wave tile 32x32
// (4 accumulators, 8 WMMAs per stage). Double-buffered LDS filled by
// global_load_async_to_lds_b128; one barrier per K stage.
// =====================================================================
#define BM 128
#define BN 64
#define BK 64

__global__ __launch_bounds__(256) void gemm_bf16_wmma(
    const unsigned short* __restrict__ A, const unsigned short* __restrict__ W,
    float* __restrict__ C, int M, int N, int K,
    const float* __restrict__ row_scale)
{
  __shared__ __align__(128) unsigned short sA[2][BM][BK];  // 2 x 16 KB
  __shared__ __align__(128) unsigned short sB[2][BN][BK];  // 2 x  8 KB

  const int tid  = threadIdx.x;
  const int wave = tid >> 5;
  const int lane = tid & 31;
  const int m0 = blockIdx.y * BM;
  const int n0 = blockIdx.x * BN;
  const int mw = (wave & 3) * 32;   // wave M offset (4 waves cover 128)
  const int nw = (wave >> 2) * 32;  // wave N offset (2 waves cover 64)
  const int hl = lane >> 4;         // lane half selects K-group
  const int lr = lane & 15;

  v8f acc00 = {}, acc01 = {}, acc10 = {}, acc11 = {};  // [mi][ni]

  auto stage = [&](int b, int kk) {
    // A tile: 128 rows x 8 chunks of 16B -> 4 chunks per thread
#pragma unroll
    for (int i = 0; i < 4; ++i) {
      const int id = tid + 256 * i;
      const int r = id >> 3, ch = (id & 7) * 8;
      async_copy_b128(A + (size_t)(m0 + r) * K + kk + ch, &sA[b][r][ch]);
    }
    // B tile: 64 rows x 8 chunks -> 2 chunks per thread
#pragma unroll
    for (int i = 0; i < 2; ++i) {
      const int id = tid + 256 * i;
      const int r = id >> 3, ch = (id & 7) * 8;
      async_copy_b128(W + (size_t)(n0 + r) * K + kk + ch, &sB[b][r][ch]);
    }
  };

  int buf = 0;
  stage(0, 0);
  wait_async_lds();
  __syncthreads();

  for (int k0 = 0; k0 < K; k0 += BK) {
    if (k0 + BK < K) stage(buf ^ 1, k0 + BK);   // overlap next stage with WMMA

#pragma unroll
    for (int ks = 0; ks < BK; ks += 32) {
      // A fragments (16x32 bf16): lanes 0-15 K=0..15, lanes 16-31 K=16..31
      v8bf a0lo = *(const v8bf*)&sA[buf][mw + lr][ks + hl * 16];
      v8bf a0hi = *(const v8bf*)&sA[buf][mw + lr][ks + hl * 16 + 8];
      v8bf a1lo = *(const v8bf*)&sA[buf][mw + 16 + lr][ks + hl * 16];
      v8bf a1hi = *(const v8bf*)&sA[buf][mw + 16 + lr][ks + hl * 16 + 8];
      v16bf a0, a1;
#pragma unroll
      for (int i = 0; i < 8; ++i) {
        a0[i] = a0lo[i]; a0[i + 8] = a0hi[i];
        a1[i] = a1lo[i]; a1[i + 8] = a1hi[i];
      }
      // B fragments (32x16 bf16): lane holds 16 contiguous K of its column
      v16bf b0 = *(const v16bf*)&sB[buf][nw + lr][ks + hl * 16];
      v16bf b1 = *(const v16bf*)&sB[buf][nw + 16 + lr][ks + hl * 16];

      acc00 = __builtin_amdgcn_wmma_f32_16x16x32_bf16(false, a0, false, b0, (short)0, acc00, false, false);
      acc01 = __builtin_amdgcn_wmma_f32_16x16x32_bf16(false, a0, false, b1, (short)0, acc01, false, false);
      acc10 = __builtin_amdgcn_wmma_f32_16x16x32_bf16(false, a1, false, b0, (short)0, acc10, false, false);
      acc11 = __builtin_amdgcn_wmma_f32_16x16x32_bf16(false, a1, false, b1, (short)0, acc11, false, false);
    }

    wait_async_lds();   // our next-stage loads have landed
    __syncthreads();    // everyone done computing on buf
    buf ^= 1;
  }

  // C/D layout: VGPR r, lanes 0-15 -> (M=r, N=lane); lanes 16-31 -> (M=r+8, N=lane-16)
  const int cn  = n0 + nw + lr;
  const int cm0 = m0 + mw + hl * 8;
#pragma unroll
  for (int r = 0; r < 8; ++r) {
    const int cmA = cm0 + r;
    const int cmB = cm0 + 16 + r;
    const float scA = row_scale ? row_scale[cmA] : 1.0f;
    const float scB = row_scale ? row_scale[cmB] : 1.0f;
    C[(size_t)cmA * N + cn]      = acc00[r] * scA;
    C[(size_t)cmA * N + cn + 16] = acc01[r] * scA;
    C[(size_t)cmB * N + cn]      = acc10[r] * scB;
    C[(size_t)cmB * N + cn + 16] = acc11[r] * scB;
  }
}

// =====================================================================
// el/er attention dots: one wave per (node, head), shuffle reduction
// =====================================================================
__global__ __launch_bounds__(128) void attn_dots(
    const float* __restrict__ fs, const float* __restrict__ attn_l,
    const float* __restrict__ attn_r, float* __restrict__ el,
    float* __restrict__ er, int ndst)
{
  const int node = blockIdx.x;
  const int h    = threadIdx.x >> 5;
  const int lane = threadIdx.x & 31;
  const float* row = fs + ((size_t)node * NH + h) * HIDD;
  float sl = 0.f, sr = 0.f;
#pragma unroll
  for (int d = lane; d < HIDD; d += 32) {
    const float v = row[d];
    sl += v * attn_l[h * HIDD + d];
    sr += v * attn_r[h * HIDD + d];
  }
#pragma unroll
  for (int off = 16; off > 0; off >>= 1) {
    sl += __shfl_down(sl, off, 32);
    sr += __shfl_down(sr, off, 32);
  }
  if (lane == 0) {
    el[node * NH + h] = sl;
    if (node < ndst) er[node * NH + h] = sr;
  }
}

// =====================================================================
// edge softmax helpers
// =====================================================================
__device__ __forceinline__ void atomicMaxF(float* addr, float val) {
  if (val >= 0.f) atomicMax((int*)addr, __float_as_int(val));
  else            atomicMin((unsigned int*)addr, __float_as_uint(val));
}

__global__ void init_md(float* __restrict__ mx, float* __restrict__ den, int n) {
  int i = blockIdx.x * blockDim.x + threadIdx.x;
  if (i < n) { mx[i] = -INFINITY; den[i] = 0.f; }
}

__global__ void edge_logit(const int* __restrict__ src, const int* __restrict__ dst,
                           const float* __restrict__ el, const float* __restrict__ er,
                           float* __restrict__ ebuf, float* __restrict__ mx, int nE)
{
  int e = blockIdx.x * blockDim.x + threadIdx.x;
  if (e >= nE) return;
  const int s = src[e], d = dst[e];
#pragma unroll
  for (int h = 0; h < NH; ++h) {
    float x = el[s * NH + h] + er[d * NH + h];
    x = (x > 0.f) ? x : NEG_SLOPE * x;                 // leaky_relu
    ebuf[(size_t)e * NH + h] = x;
    atomicMaxF(&mx[d * NH + h], x);
  }
}

__global__ void edge_exp(const int* __restrict__ dst, const float* __restrict__ mx,
                         float* __restrict__ ebuf, float* __restrict__ den, int nE)
{
  int e = blockIdx.x * blockDim.x + threadIdx.x;
  if (e >= nE) return;
  const int d = dst[e];
#pragma unroll
  for (int h = 0; h < NH; ++h) {
    const float ex = __expf(ebuf[(size_t)e * NH + h] - mx[d * NH + h]);
    ebuf[(size_t)e * NH + h] = ex;
    atomicAdd(&den[d * NH + h], ex);
  }
}

// one block per edge: rst[dst] += a[h] * feat_src[src]  (512 f32 atomics)
__global__ __launch_bounds__(256) void edge_scatter(
    const int* __restrict__ src, const int* __restrict__ dst,
    const float* __restrict__ ebuf, const float* __restrict__ den,
    const float* __restrict__ fs, float* __restrict__ rst)
{
  const int e = blockIdx.x;
  const int s = src[e], d = dst[e];
  __shared__ float a[NH];
  if (threadIdx.x < NH)
    a[threadIdx.x] = ebuf[(size_t)e * NH + threadIdx.x] /
                     fmaxf(den[d * NH + threadIdx.x], 1e-9f);
  __syncthreads();
  const float* fsr = fs  + (size_t)s * (NH * HIDD);
  float*       out = rst + (size_t)d * (NH * HIDD);
#pragma unroll
  for (int i = threadIdx.x; i < NH * HIDD; i += 256)
    atomicAdd(&out[i], a[i >> 7] * fsr[i]);
}

// =====================================================================
// masked coefficient sum: s[n] = sum_{j < min(cnt,50)} poi_coeff[nid, j]
// =====================================================================
__global__ void coeff_sum(const int* __restrict__ out_nodes,
                          const int* __restrict__ counts,
                          const float* __restrict__ coeff,
                          float* __restrict__ s, int n)
{
  int i = blockIdx.x * blockDim.x + threadIdx.x;
  if (i >= n) return;
  const int nid = out_nodes[i];
  int c = counts[nid]; c = (c > KCAP) ? KCAP : c;
  const float* cp = coeff + (size_t)nid * KCAP;
  float acc = 0.f;
  for (int j = 0; j < c; ++j) acc += cp[j];
  s[i] = acc;
}

// =====================================================================
// finalize: out = relu((elu(rst + bias) + fdist_broadcast) * 0.5)
// =====================================================================
__global__ void finalize(const float* __restrict__ rst, const float* __restrict__ bias,
                         const float* __restrict__ fdist, float* __restrict__ out, int total)
{
  int i = blockIdx.x * blockDim.x + threadIdx.x;
  if (i >= total) return;
  const int n  = i >> 9;          // / 512
  const int hd = i & 511;
  const int d  = hd & (HIDD - 1);
  float x = rst[i] + bias[hd];
  x = (x > 0.f) ? x : (__expf(x) - 1.f);            // elu
  const float v = 0.5f * (x + fdist[(size_t)n * HIDD + d]);
  out[i] = fmaxf(v, 0.f);
}

// =====================================================================
// launcher
// =====================================================================
extern "C" void kernel_launch(void* const* d_in, const int* in_sizes, int n_in,
                              void* d_out, int out_size, void* d_ws, size_t ws_size,
                              hipStream_t stream) {
  const float* feat      = (const float*)d_in[0];
  const float* poi_cat   = (const float*)d_in[1];
  const float* poi_coeff = (const float*)d_in[2];
  const float* fc_w      = (const float*)d_in[3];
  const float* attn_l    = (const float*)d_in[4];
  const float* attn_r    = (const float*)d_in[5];
  const float* bias_p    = (const float*)d_in[6];
  const float* res_w     = (const float*)d_in[7];
  const float* w_w       = (const float*)d_in[8];
  const int*   src_idx   = (const int*)d_in[9];
  const int*   dst_idx   = (const int*)d_in[10];
  const int*   out_nodes = (const int*)d_in[11];
  const int*   nbr_cnt   = (const int*)d_in[12];
  float* out = (float*)d_out;

  float* p = (float*)d_ws;
  float* feat_src = p;  p += (size_t)N_SRC * NH * HIDD;   // 128 MB
  float* el    = p;     p += (size_t)N_SRC * NH;
  float* er    = p;     p += (size_t)N_DST * NH;
  float* mx    = p;     p += (size_t)N_DST * NH;
  float* den   = p;     p += (size_t)N_DST * NH;
  float* ebuf  = p;     p += (size_t)NE * NH;
  float* rst   = p;     p += (size_t)N_DST * NH * HIDD;   // residual + messages
  float* fdist = p;     p += (size_t)N_DST * HIDD;
  float* sbuf  = p;     p += (size_t)N_DST;
  unsigned short* q = (unsigned short*)p;
  unsigned short* feat_bf = q; q += (size_t)N_SRC * DIN;  // 32 MB
  unsigned short* fcw_bf  = q; q += (size_t)(NH * HIDD) * DIN;
  unsigned short* resw_bf = q; q += (size_t)(NH * HIDD) * DIN;
  unsigned short* ww_bf   = q; q += (size_t)HIDD * DIN;
  unsigned short* cf_bf   = q; q += (size_t)N_DST * DIN;  // gathered poi_cat rows

  const int CPR = DIN / 8;   // 16B chunks per row

  // 0) one-time bf16 conversions (also folds the poi_cat gather)
  f32_to_bf16_rows<<<(N_SRC * CPR + 255) / 256, 256, 0, stream>>>(feat, feat_bf, N_SRC, DIN, nullptr);
  f32_to_bf16_rows<<<(NH * HIDD * CPR + 255) / 256, 256, 0, stream>>>(fc_w, fcw_bf, NH * HIDD, DIN, nullptr);
  f32_to_bf16_rows<<<(NH * HIDD * CPR + 255) / 256, 256, 0, stream>>>(res_w, resw_bf, NH * HIDD, DIN, nullptr);
  f32_to_bf16_rows<<<(HIDD * CPR + 255) / 256, 256, 0, stream>>>(w_w, ww_bf, HIDD, DIN, nullptr);
  f32_to_bf16_rows<<<(N_DST * CPR + 255) / 256, 256, 0, stream>>>(poi_cat, cf_bf, N_DST, DIN, out_nodes);

  // 1) feat_src = feat @ fc_w^T          [65536,512]
  gemm_bf16_wmma<<<dim3((NH * HIDD) / BN, N_SRC / BM), 256, 0, stream>>>(
      feat_bf, fcw_bf, feat_src, N_SRC, NH * HIDD, DIN, nullptr);
  // 2) attention dots
  attn_dots<<<N_SRC, 128, 0, stream>>>(feat_src, attn_l, attn_r, el, er, N_DST);
  // 3) init segment max / denom
  init_md<<<(N_DST * NH + 255) / 256, 256, 0, stream>>>(mx, den, N_DST * NH);
  // 4) logits + segment max
  edge_logit<<<(NE + 255) / 256, 256, 0, stream>>>(src_idx, dst_idx, el, er, ebuf, mx, NE);
  // 5) exp + segment sum
  edge_exp<<<(NE + 255) / 256, 256, 0, stream>>>(dst_idx, mx, ebuf, den, NE);
  // 6) residual projection directly into rst  [16384,512]
  gemm_bf16_wmma<<<dim3((NH * HIDD) / BN, N_DST / BM), 256, 0, stream>>>(
      feat_bf, resw_bf, rst, N_DST, NH * HIDD, DIN, nullptr);
  // 7) scatter weighted messages onto rst
  edge_scatter<<<NE, 256, 0, stream>>>(src_idx, dst_idx, ebuf, den, feat_src, rst);
  // 8) masked coefficient sums
  coeff_sum<<<(N_DST + 255) / 256, 256, 0, stream>>>(out_nodes, nbr_cnt, poi_coeff, sbuf, N_DST);
  // 9) fdist = cf_bf @ w_w^T, row-scaled by s   [16384,128]
  gemm_bf16_wmma<<<dim3(HIDD / BN, N_DST / BM), 256, 0, stream>>>(
      cf_bf, ww_bf, fdist, N_DST, HIDD, DIN, sbuf);
  // 10) fused elu + relu epilogue
  finalize<<<(N_DST * NH * HIDD + 255) / 256, 256, 0, stream>>>(
      rst, bias_p, fdist, out, N_DST * NH * HIDD);
}
